// MultiHeadAttention_63677185130929
// MI455X (gfx1250) — matrix operand order repaired and checked by reading, not compiled, not run on hardware
//
#include <hip/hip_runtime.h>
#include <stdint.h>

// ---------------------------------------------------------------------------
// MHA: B=2, S=2048, E=1024, H=16, d=64.  bf16 WMMA compute, fp32 accumulate.
// ---------------------------------------------------------------------------

typedef __attribute__((ext_vector_type(16))) __bf16       v16bf;
typedef __attribute__((ext_vector_type(8)))  float        v8f;
typedef __attribute__((ext_vector_type(4)))  unsigned int v4u;

union BF16x16 { v16bf v; v4u u[2]; };

static __device__ __forceinline__ unsigned short f2bf(float f) {
  unsigned int u = __float_as_uint(f);
  u += 0x7fffu + ((u >> 16) & 1u);       // round-to-nearest-even
  return (unsigned short)(u >> 16);
}

static __device__ __forceinline__ v8f wmma_bf16(v16bf a, v16bf b, v8f c) {
  return __builtin_amdgcn_wmma_f32_16x16x32_bf16(false, a, false, b,
                                                 (short)0, c, false, false);
}

static __device__ __forceinline__ v8f vzero8() {
  v8f z = {0.f, 0.f, 0.f, 0.f, 0.f, 0.f, 0.f, 0.f};
  return z;
}

// ---------------------------------------------------------------------------
// fp32 -> bf16 elementwise convert
// ---------------------------------------------------------------------------
__global__ __launch_bounds__(256)
void f32_to_bf16_kernel(const float* __restrict__ in,
                        unsigned short* __restrict__ out, int n) {
  for (int i = blockIdx.x * 256 + threadIdx.x; i < n; i += gridDim.x * 256)
    out[i] = f2bf(in[i]);
}

// ---------------------------------------------------------------------------
// QKV projection: [4096,1024] x [3072,1024]^T + b  -> Q,K (row major, Q*0.125)
// and V^T, all bf16, laid out [B,H,S,64] / [B,H,64,S].
// One wave computes a 16x64 tile (4 accumulators, K-loop of 32).
// Head mapping follows reference reshape: head = c/192, part=(c%192)/64.
// ---------------------------------------------------------------------------
__global__ __launch_bounds__(256)
void qkv_gemm_kernel(const unsigned short* __restrict__ xb,
                     const unsigned short* __restrict__ wb,
                     const float* __restrict__ bias,
                     unsigned short* __restrict__ qbuf,
                     unsigned short* __restrict__ kbuf,
                     unsigned short* __restrict__ vtbuf) {
  const int lane = threadIdx.x & 31;
  const int wave = threadIdx.x >> 5;
  const int wid  = blockIdx.x * 8 + wave;   // 12288 waves
  const int mt   = wid / 48;                // 0..255  (row tile)
  const int nc   = wid % 48;                // 0..47   (64-wide col chunk)
  const int hi   = lane >> 4;
  const int ln   = lane & 15;

  const v4u* ap = (const v4u*)(xb + (size_t)(mt * 16 + ln) * 1024);
  const v4u* bp[4];
#pragma unroll
  for (int t = 0; t < 4; ++t)
    bp[t] = (const v4u*)(wb + (size_t)(nc * 64 + t * 16 + ln) * 1024);

  v8f acc[4];
#pragma unroll
  for (int t = 0; t < 4; ++t) acc[t] = vzero8();

  for (int k0 = 0; k0 < 1024; k0 += 32) {
    const int ka = k0 >> 3;
    __builtin_prefetch(ap + ka + 8, 0, 1);          // global_prefetch_b8
    BF16x16 a;
    a.u[0] = ap[ka + 2 * hi];                       // K = 16*hi .. 16*hi+7
    a.u[1] = ap[ka + 2 * hi + 1];                   // K = 16*hi+8 .. 16*hi+15
#pragma unroll
    for (int t = 0; t < 4; ++t) {
      BF16x16 bm;
      bm.u[0] = bp[t][ka + 2 * hi];                 // K = 16*hi .. 16*hi+15
      bm.u[1] = bp[t][ka + 2 * hi + 1];
      acc[t] = wmma_bf16(a.v, bm.v, acc[t]);
    }
  }

  const int m0 = mt * 16 + 8 * hi;
#pragma unroll
  for (int t = 0; t < 4; ++t) {
    const int ncol = nc * 64 + t * 16 + ln;
    const float bv = bias[ncol];
    const int head = ncol / 192;        // reference reshape: (B,S,H,3d)
    const int part = (ncol % 192) >> 6; // 0=Q 1=K 2=V  (uniform per tile)
    const int dcol = ncol & 63;
#pragma unroll
    for (int r = 0; r < 8; ++r) {
      const int m  = m0 + r;            // global row in [0,4096)
      const int bb = m >> 11;
      const int s  = m & 2047;
      const float val = acc[t][r] + bv;
      if (part == 0)
        qbuf[((size_t)((bb * 16 + head) * 2048 + s)) * 64 + dcol] =
            f2bf(val * 0.125f);         // pre-scale Q by 1/sqrt(64)
      else if (part == 1)
        kbuf[((size_t)((bb * 16 + head) * 2048 + s)) * 64 + dcol] = f2bf(val);
      else
        vtbuf[((size_t)((bb * 16 + head) * 64 + dcol)) * 2048 + s] = f2bf(val);
    }
  }
}

// ---------------------------------------------------------------------------
// Flash attention: one wave per (b,h,q-tile of 16). 32 keys per iteration:
// 4 WMMA for scores, online softmax, P->LDS->A-fragment, 4 WMMA for PV.
// ---------------------------------------------------------------------------
__global__ __launch_bounds__(32)
void attn_kernel(const unsigned short* __restrict__ qbuf,
                 const unsigned short* __restrict__ kbuf,
                 const unsigned short* __restrict__ vtbuf,
                 unsigned short* __restrict__ attnb) {
  __shared__ __align__(16) unsigned short pls[16 * 32];   // P tile, bf16

  const int lane = threadIdx.x;
  const int hi = lane >> 4, ln = lane & 15;
  const int bh = blockIdx.x;        // b*16 + h
  const int qt = blockIdx.y;        // 0..127
  const int b = bh >> 4, h = bh & 15;

  const unsigned short* qp = qbuf + (size_t)bh * 2048 * 64;
  const unsigned short* kp = kbuf + (size_t)bh * 2048 * 64;
  const unsigned short* vp = vtbuf + (size_t)bh * 64 * 2048;

  // Q A-fragments for K-dim d=64 (two 16x32 slabs), loaded once.
  BF16x16 qa0, qa1;
  {
    const v4u* q4 = (const v4u*)(qp + (size_t)(qt * 16 + ln) * 64);
    qa0.u[0] = q4[2 * hi];     qa0.u[1] = q4[2 * hi + 1];
    qa1.u[0] = q4[4 + 2 * hi]; qa1.u[1] = q4[5 + 2 * hi];
  }

  v8f o[4];
#pragma unroll
  for (int t = 0; t < 4; ++t) o[t] = vzero8();
  float mrow[8], lrow[8];
#pragma unroll
  for (int r = 0; r < 8; ++r) { mrow[r] = -1e30f; lrow[r] = 0.f; }

  for (int j0 = 0; j0 < 2048; j0 += 32) {
    // ---- scores: S[16q x 32k] = Q(16x64) . K^T ----
    v8f s1 = vzero8(), s2 = vzero8();
    {
      const v4u* k4 = (const v4u*)(kp + (size_t)(j0 + ln) * 64);
      BF16x16 bk;
      bk.u[0] = k4[2 * hi];     bk.u[1] = k4[2 * hi + 1];
      s1 = wmma_bf16(qa0.v, bk.v, s1);
      bk.u[0] = k4[4 + 2 * hi]; bk.u[1] = k4[5 + 2 * hi];
      s1 = wmma_bf16(qa1.v, bk.v, s1);
    }
    {
      const v4u* k4 = (const v4u*)(kp + (size_t)(j0 + 16 + ln) * 64);
      BF16x16 bk;
      bk.u[0] = k4[2 * hi];     bk.u[1] = k4[2 * hi + 1];
      s2 = wmma_bf16(qa0.v, bk.v, s2);
      bk.u[0] = k4[4 + 2 * hi]; bk.u[1] = k4[5 + 2 * hi];
      s2 = wmma_bf16(qa1.v, bk.v, s2);
    }

    // ---- online softmax (rows live across 16-lane halves) ----
    v8f p1, p2;
    float corr[8];
#pragma unroll
    for (int r = 0; r < 8; ++r) {
      float tmax = fmaxf(s1[r], s2[r]);
      tmax = fmaxf(tmax, __shfl_xor(tmax, 1));
      tmax = fmaxf(tmax, __shfl_xor(tmax, 2));
      tmax = fmaxf(tmax, __shfl_xor(tmax, 4));
      tmax = fmaxf(tmax, __shfl_xor(tmax, 8));
      const float mn = fmaxf(mrow[r], tmax);
      corr[r] = __expf(mrow[r] - mn);
      const float e1 = __expf(s1[r] - mn);
      const float e2 = __expf(s2[r] - mn);
      p1[r] = e1; p2[r] = e2;
      float rs = e1 + e2;
      rs += __shfl_xor(rs, 1);
      rs += __shfl_xor(rs, 2);
      rs += __shfl_xor(rs, 4);
      rs += __shfl_xor(rs, 8);
      lrow[r] = lrow[r] * corr[r] + rs;
      mrow[r] = mn;
    }
#pragma unroll
    for (int t = 0; t < 4; ++t)
#pragma unroll
      for (int r = 0; r < 8; ++r) o[t][r] *= corr[r];

    // ---- C-layout P -> A-fragment via LDS ----
    __syncthreads();
#pragma unroll
    for (int r = 0; r < 8; ++r) {
      pls[(r + 8 * hi) * 32 + ln]      = f2bf(p1[r]);
      pls[(r + 8 * hi) * 32 + 16 + ln] = f2bf(p2[r]);
    }
    __syncthreads();
    BF16x16 pa;
    {
      const v4u* pp = (const v4u*)pls;
      pa.u[0] = pp[ln * 4 + 2 * hi];
      pa.u[1] = pp[ln * 4 + 2 * hi + 1];
    }

    // ---- O += P(16x32) . V(32x64), B-fragments from V^T rows ----
#pragma unroll
    for (int t = 0; t < 4; ++t) {
      const v4u* v4 = (const v4u*)(vp + (size_t)(t * 16 + ln) * 2048 + j0);
      BF16x16 bv;
      bv.u[0] = v4[2 * hi]; bv.u[1] = v4[2 * hi + 1];
      o[t] = wmma_bf16(pa.v, bv.v, o[t]);
    }
  }

  // ---- normalize and store bf16 attn output in [B*S, E] (head-merged) ----
  float inv[8];
#pragma unroll
  for (int r = 0; r < 8; ++r) inv[r] = 1.0f / lrow[r];
  const size_t rowbase = (size_t)(b * 2048 + qt * 16) * 1024 + h * 64;
#pragma unroll
  for (int t = 0; t < 4; ++t)
#pragma unroll
    for (int r = 0; r < 8; ++r)
      attnb[rowbase + (size_t)(r + 8 * hi) * 1024 + t * 16 + ln] =
          f2bf(o[t][r] * inv[r]);
}

// ---------------------------------------------------------------------------
// Output projection: attn[4096,1024] x out_w[1024,1024]^T + b -> fp32 out
// ---------------------------------------------------------------------------
__global__ __launch_bounds__(256)
void out_gemm_kernel(const unsigned short* __restrict__ ab,
                     const unsigned short* __restrict__ wb,
                     const float* __restrict__ bias,
                     float* __restrict__ out) {
  const int lane = threadIdx.x & 31;
  const int wave = threadIdx.x >> 5;
  const int wid  = blockIdx.x * 8 + wave;   // 4096 waves
  const int mt   = wid >> 4;                // 0..255
  const int nc   = wid & 15;                // 0..15
  const int hi   = lane >> 4;
  const int ln   = lane & 15;

  const v4u* ap = (const v4u*)(ab + (size_t)(mt * 16 + ln) * 1024);
  const v4u* bp[4];
#pragma unroll
  for (int t = 0; t < 4; ++t)
    bp[t] = (const v4u*)(wb + (size_t)(nc * 64 + t * 16 + ln) * 1024);

  v8f acc[4];
#pragma unroll
  for (int t = 0; t < 4; ++t) acc[t] = vzero8();

  for (int k0 = 0; k0 < 1024; k0 += 32) {
    const int ka = k0 >> 3;
    __builtin_prefetch(ap + ka + 8, 0, 1);
    BF16x16 a;
    a.u[0] = ap[ka + 2 * hi];
    a.u[1] = ap[ka + 2 * hi + 1];
#pragma unroll
    for (int t = 0; t < 4; ++t) {
      BF16x16 bm;
      bm.u[0] = bp[t][ka + 2 * hi];
      bm.u[1] = bp[t][ka + 2 * hi + 1];
      acc[t] = wmma_bf16(a.v, bm.v, acc[t]);
    }
  }

  const int m0 = mt * 16 + 8 * hi;
#pragma unroll
  for (int t = 0; t < 4; ++t) {
    const int ncol = nc * 64 + t * 16 + ln;
    const float bv = bias[ncol];
#pragma unroll
    for (int r = 0; r < 8; ++r)
      out[(size_t)(m0 + r) * 1024 + ncol] = acc[t][r] + bv;
  }
}

// ---------------------------------------------------------------------------
// launch
// ---------------------------------------------------------------------------
extern "C" void kernel_launch(void* const* d_in, const int* in_sizes, int n_in,
                              void* d_out, int out_size, void* d_ws,
                              size_t ws_size, hipStream_t stream) {
  const float* x     = (const float*)d_in[0];
  const float* qkv_w = (const float*)d_in[1];
  const float* qkv_b = (const float*)d_in[2];
  const float* out_w = (const float*)d_in[3];
  const float* out_b = (const float*)d_in[4];
  float* out = (float*)d_out;

  // workspace layout (bf16 elements), ~48 MB total
  unsigned short* ws    = (unsigned short*)d_ws;
  unsigned short* xb    = ws;                 // 4096*1024
  unsigned short* qkvwb = xb + 4194304;       // 3072*1024
  unsigned short* outwb = qkvwb + 3145728;    // 1024*1024
  unsigned short* qbuf  = outwb + 1048576;    // [B,H,S,64]
  unsigned short* kbuf  = qbuf + 4194304;     // [B,H,S,64]
  unsigned short* vtbuf = kbuf + 4194304;     // [B,H,64,S]
  unsigned short* attnb = vtbuf + 4194304;    // [B*S, E]

  f32_to_bf16_kernel<<<2048, 256, 0, stream>>>(x, xb, 4194304);
  f32_to_bf16_kernel<<<2048, 256, 0, stream>>>(qkv_w, qkvwb, 3145728);
  f32_to_bf16_kernel<<<1024, 256, 0, stream>>>(out_w, outwb, 1048576);

  qkv_gemm_kernel<<<1536, 256, 0, stream>>>(xb, qkvwb, qkv_b,
                                            qbuf, kbuf, vtbuf);

  attn_kernel<<<dim3(32, 128), 32, 0, stream>>>(qbuf, kbuf, vtbuf, attnb);

  out_gemm_kernel<<<512, 256, 0, stream>>>(attnb, outwb, out_b, out);
}
